// RWKVTimeMix_45509473468953
// MI455X (gfx1250) — matrix operand order repaired and checked by reading, not compiled, not run on hardware
//
#include <hip/hip_runtime.h>

// ---------------------------------------------------------------------------
// RWKV TimeMix for MI455X (gfx1250, wave32, WMMA + TDM async tensor loads).
// B=4, T=8192, C=1024.  GEMMs in bf16 WMMA (f32 accum), scan in f32.
// ---------------------------------------------------------------------------

typedef __attribute__((ext_vector_type(16))) __bf16 v16bf;
typedef __attribute__((ext_vector_type(8)))  float  v8f;
typedef __attribute__((ext_vector_type(4)))  __bf16 v4bf;
typedef __attribute__((ext_vector_type(4)))  unsigned int v4u;
typedef __attribute__((ext_vector_type(8)))  int v8i;
typedef __attribute__((ext_vector_type(4)))  int v4i;

#define BSZ   4
#define TLEN  8192
#define CDIM  1024
#define MROWS (BSZ * TLEN)   // 32768

// LDS tile: 128 rows x 64 bf16 (128B) + 16B pad per row -> 144B row stride
#define ROWB  144
#define TILEB (128 * ROWB)   // 18432 B

union FragU {
  v16bf v;
  uint4 u[2];
};

// ---------------------------------------------------------------------------
// TDM: issue one 2D tile load (128 rows x 64 bf16) global -> LDS.
// Descriptor per CDNA5 ISA 8.3-8.6:  data_size=2B, pad 4 dwords / 32 dwords.
// ---------------------------------------------------------------------------
__device__ __forceinline__ void tdm_load_tile_2d(unsigned lds_byte_addr,
                                                 const void* gtile,
                                                 unsigned tensor_rows) {
  const unsigned long long ga = (unsigned long long)gtile;
  v4u g0;
  g0[0] = 1u;                                              // count=1, user mode
  g0[1] = lds_byte_addr;                                   // lds_addr
  g0[2] = (unsigned)ga;                                    // global_addr[31:0]
  g0[3] = ((unsigned)(ga >> 32) & 0x01ffffffu) | (2u << 30);  // addr[56:32]|type=2

  v8i g1;
  g1[0] = (int)((1u << 16)      // data_size = 1 -> 2 bytes
              | (1u << 20)      // pad_enable
              | (4u << 22)      // pad_interval: 32 dwords (one 128B row)
              | (3u << 25));    // pad_amount: 4 dwords (16B)
  g1[1] = (int)((CDIM & 0xffffu) << 16);                   // tensor_dim0 lo16
  g1[2] = (int)((CDIM >> 16) | ((tensor_rows & 0xffffu) << 16)); // dim0 hi | dim1 lo
  g1[3] = (int)((tensor_rows >> 16) | (64u << 16));        // dim1 hi | tile_dim0=64
  g1[4] = (int)(128u);                                     // tile_dim1=128, tile_dim2=0
  g1[5] = (int)CDIM;                                       // tensor_dim0_stride lo32
  g1[6] = 0;                                               // stride hi | dim1_stride lo
  g1[7] = 0;

  v4i g2; g2[0] = 1; g2[1] = 1; g2[2] = 0; g2[3] = 0;      // dims 2/3 unused
  v4i g3; g3[0] = 0; g3[1] = (int)(1u << 16); g3[2] = 0; g3[3] = 0;

#if defined(__clang_major__) && (__clang_major__ >= 23)
  v8i g4 = {};
  __builtin_amdgcn_tensor_load_to_lds(g0, g1, g2, g3, g4, 0);
#else
  __builtin_amdgcn_tensor_load_to_lds(g0, g1, g2, g3, 0);
#endif
}

// A fragment (16x32 bf16, MxK): lane = 16*khalf + m.
// From LDS tile row: bytes [cb + 16*kh, +16) and [cb + 16*kh + 32, +16).
__device__ __forceinline__ v16bf lds_frag_a(const char* base, int row, int cb,
                                            int kh) {
  FragU f;
  const char* p = base + row * ROWB + cb + kh * 16;
  f.u[0] = *reinterpret_cast<const uint4*>(p);
  f.u[1] = *reinterpret_cast<const uint4*>(p + 32);
  return f.v;
}

// B fragment (32x16 bf16, KxN), B[k,n] = W[n,k]: lane holds W row n,
// bytes [cb + 16*kh, +16) and [cb + 16*kh + 32, +16).
__device__ __forceinline__ v16bf lds_frag_b(const char* base, int row, int cb,
                                            int kh) {
  FragU f;
  const char* p = base + row * ROWB + cb + kh * 16;
  f.u[0] = *reinterpret_cast<const uint4*>(p);
  f.u[1] = *reinterpret_cast<const uint4*>(p + 32);
  return f.v;
}

// C/D 16x16 f32 layout: VGPR r holds row (8*khalf + r), col = lane%16.
__device__ __forceinline__ void store_acc(float* __restrict__ C, int mbase,
                                          int col, v8f acc) {
  float* p = C + (size_t)mbase * CDIM + col;
#pragma unroll
  for (int r = 0; r < 8; ++r) p[(size_t)r * CDIM] = acc[r];
}

// ---------------------------------------------------------------------------
// GEMM:  C[M,N] = A[M,K] * W[N,K]^T   (nn.Linear), M=32768, N=K=1024.
// 8 waves / block own a 128x128 tile; K-step 64; TDM double-buffered LDS.
// ---------------------------------------------------------------------------
__global__ __launch_bounds__(256) void gemm_bf16_kernel(
    const __bf16* __restrict__ A, const __bf16* __restrict__ W,
    float* __restrict__ C) {
  constexpr int K = CDIM;
  __shared__ __attribute__((aligned(128))) char smem[4 * TILEB];

  const int lane   = threadIdx.x & 31;
  const int wv     = threadIdx.x >> 5;       // 0..7
  const int msub   = (wv & 3) * 32;          // wave M offset within 128 tile
  const int nsub   = (wv >> 2) * 64;         // wave N offset within 128 tile
  const int blockN = blockIdx.x & 7;         // N/128 = 8
  const int blockM = blockIdx.x >> 3;        // M/128 = 256
  const int m0 = blockM * 128;
  const int n0 = blockN * 128;
  const int lm = lane & 15;
  const int kh = lane >> 4;

  const unsigned ldsbase = (unsigned)(unsigned long long)(void*)smem;
  const bool isW0 = (wv == 0);

  // Prologue: k=0 tiles into buffer 0 (A tiles in buf 0/1, B tiles in 2/3).
  if (isW0) {
    tdm_load_tile_2d(ldsbase + 0 * TILEB, A + (size_t)m0 * K, MROWS);
    tdm_load_tile_2d(ldsbase + 2 * TILEB, W + (size_t)n0 * K, CDIM);
  }

  v8f acc[2][4] = {};

  constexpr int NSTEP = K / 64;  // 16
  for (int s = 0; s < NSTEP; ++s) {
    if (isW0) {
      if (s + 1 < NSTEP) {
        const int kn = (s + 1) * 64;
        tdm_load_tile_2d(ldsbase + ((s + 1) & 1) * TILEB,
                         A + (size_t)m0 * K + kn, MROWS);
        tdm_load_tile_2d(ldsbase + (2 + ((s + 1) & 1)) * TILEB,
                         W + (size_t)n0 * K + kn, CDIM);
        __builtin_amdgcn_s_wait_tensorcnt(2);  // current pair done, next in flight
      } else {
        __builtin_amdgcn_s_wait_tensorcnt(0);
      }
    }
    __syncthreads();

    const char* Ac = smem + (s & 1) * TILEB;
    const char* Bc = smem + (2 + (s & 1)) * TILEB;
#pragma unroll
    for (int ks = 0; ks < 2; ++ks) {      // two 32-K substeps of the 64-K tile
      const int cb = ks * 64;             // byte offset of K-chunk within row
      v16bf a0 = lds_frag_a(Ac, msub + lm, cb, kh);
      v16bf a1 = lds_frag_a(Ac, msub + 16 + lm, cb, kh);
#pragma unroll
      for (int j = 0; j < 4; ++j) {
        v16bf b = lds_frag_b(Bc, nsub + 16 * j + lm, cb, kh);
        acc[0][j] = __builtin_amdgcn_wmma_f32_16x16x32_bf16(
            false, a0, false, b, (short)0, acc[0][j], false, false);
        acc[1][j] = __builtin_amdgcn_wmma_f32_16x16x32_bf16(
            false, a1, false, b, (short)0, acc[1][j], false, false);
      }
    }
    __syncthreads();
  }

#pragma unroll
  for (int i = 0; i < 2; ++i)
#pragma unroll
    for (int j = 0; j < 4; ++j)
      store_acc(C, m0 + msub + 16 * i + 8 * kh, n0 + nsub + 16 * j, acc[i][j]);
}

// ---------------------------------------------------------------------------
// LayerNorm over C=1024 per row, output bf16. One block (256 thr) per row.
// ---------------------------------------------------------------------------
__global__ __launch_bounds__(256) void ln_kernel(
    const float* __restrict__ x, const float* __restrict__ lw,
    const float* __restrict__ lb, __bf16* __restrict__ xx) {
  const int row = blockIdx.x;
  const float4 v = reinterpret_cast<const float4*>(x + (size_t)row * CDIM)[threadIdx.x];
  float s  = v.x + v.y + v.z + v.w;
  float s2 = v.x * v.x + v.y * v.y + v.z * v.z + v.w * v.w;
#pragma unroll
  for (int off = 16; off > 0; off >>= 1) {
    s  += __shfl_xor(s, off, 32);
    s2 += __shfl_xor(s2, off, 32);
  }
  __shared__ float ssum[8], ssq[8];
  const int wv = threadIdx.x >> 5;
  if ((threadIdx.x & 31) == 0) { ssum[wv] = s; ssq[wv] = s2; }
  __syncthreads();
  float tot = 0.f, tot2 = 0.f;
#pragma unroll
  for (int i = 0; i < 8; ++i) { tot += ssum[i]; tot2 += ssq[i]; }
  const float mu  = tot * (1.0f / CDIM);
  const float var = tot2 * (1.0f / CDIM) - mu * mu;
  const float inv = rsqrtf(var + 1e-5f);
  const float4 w4 = reinterpret_cast<const float4*>(lw)[threadIdx.x];
  const float4 b4 = reinterpret_cast<const float4*>(lb)[threadIdx.x];
  v4bf o;
  o[0] = (__bf16)(((v.x - mu) * inv) * w4.x + b4.x);
  o[1] = (__bf16)(((v.y - mu) * inv) * w4.y + b4.y);
  o[2] = (__bf16)(((v.z - mu) * inv) * w4.z + b4.z);
  o[3] = (__bf16)(((v.w - mu) * inv) * w4.w + b4.w);
  reinterpret_cast<v4bf*>(xx + (size_t)row * CDIM)[threadIdx.x] = o;
}

// f32 -> bf16 weight cast, 4 elems/thread
__global__ __launch_bounds__(256) void cvt_kernel(const float* __restrict__ src,
                                                  __bf16* __restrict__ dst) {
  const int i = blockIdx.x * blockDim.x + threadIdx.x;
  const float4 v = reinterpret_cast<const float4*>(src)[i];
  v4bf o;
  o[0] = (__bf16)v.x; o[1] = (__bf16)v.y; o[2] = (__bf16)v.z; o[3] = (__bf16)v.w;
  reinterpret_cast<v4bf*>(dst)[i] = o;
}

// ---------------------------------------------------------------------------
// WKV scan: one thread per (b, c); exclusive cumsum of k*v*exp(td*t) over T,
// then y = sigmoid(r) * (hist + k*exp(tf)), stored bf16 for the final GEMM.
// ---------------------------------------------------------------------------
__global__ __launch_bounds__(256) void wkv_kernel(
    const float* __restrict__ kb, const float* __restrict__ vb,
    const float* __restrict__ rb, const float* __restrict__ tdec,
    const float* __restrict__ tfirst, __bf16* __restrict__ y) {
  const int gid = blockIdx.x * blockDim.x + threadIdx.x;  // 0..4095
  const int c = gid & (CDIM - 1);
  const int b = gid >> 10;
  const float td    = -__expf(tdec[c]);
  const float tf    = __expf(tfirst[c]);
  const float decay = __expf(td);
  float w = 1.0f, hist = 0.0f;
  size_t idx = ((size_t)b * TLEN) * CDIM + c;
  for (int t = 0; t < TLEN; ++t, idx += CDIM) {
    const float kk = kb[idx];
    const float vv = vb[idx];
    const float rr = rb[idx];
    const float wkv = hist + kk * tf;
    const float sig = 1.0f / (1.0f + __expf(-rr));
    y[idx] = (__bf16)(sig * wkv);
    hist = fmaf(kk * vv, w, hist);  // exclusive: update after use
    w *= decay;
  }
}

// ---------------------------------------------------------------------------
extern "C" void kernel_launch(void* const* d_in, const int* in_sizes, int n_in,
                              void* d_out, int out_size, void* d_ws,
                              size_t ws_size, hipStream_t stream) {
  const float* x  = (const float*)d_in[0];
  const float* td = (const float*)d_in[1];
  const float* tf = (const float*)d_in[2];
  const float* Wk = (const float*)d_in[3];
  const float* Wv = (const float*)d_in[4];
  const float* Wr = (const float*)d_in[5];
  const float* Wo = (const float*)d_in[6];
  const float* lw = (const float*)d_in[7];
  const float* lb = (const float*)d_in[8];
  float* out = (float*)d_out;

  const size_t MC = (size_t)MROWS * CDIM;  // 33,554,432
  const size_t CC = (size_t)CDIM * CDIM;   //  1,048,576

  char* ws = (char*)d_ws;
  __bf16* xx   = (__bf16*)ws;          // 64 MB
  __bf16* wk16 = xx + MC;              //  2 MB
  __bf16* wv16 = wk16 + CC;            //  2 MB
  __bf16* wr16 = wv16 + CC;            //  2 MB
  __bf16* wo16 = wr16 + CC;            //  2 MB
  float*  kbuf = (float*)(wo16 + CC);  // 128 MB
  float*  vbuf = kbuf + MC;            // 128 MB
  float*  rbuf = vbuf + MC;            // 128 MB
  __bf16* ybuf = (__bf16*)(rbuf + MC); // 64 MB  (total ~520 MB)

  ln_kernel<<<MROWS, 256, 0, stream>>>(x, lw, lb, xx);

  const int cvtBlocks = (int)(CC / 1024);  // 256 thr * 4 elem
  cvt_kernel<<<cvtBlocks, 256, 0, stream>>>(Wk, wk16);
  cvt_kernel<<<cvtBlocks, 256, 0, stream>>>(Wv, wv16);
  cvt_kernel<<<cvtBlocks, 256, 0, stream>>>(Wr, wr16);
  cvt_kernel<<<cvtBlocks, 256, 0, stream>>>(Wo, wo16);

  const int gemmBlocks = (MROWS / 128) * (CDIM / 128);  // 2048
  gemm_bf16_kernel<<<gemmBlocks, 256, 0, stream>>>(xx, wk16, kbuf);
  gemm_bf16_kernel<<<gemmBlocks, 256, 0, stream>>>(xx, wv16, vbuf);
  gemm_bf16_kernel<<<gemmBlocks, 256, 0, stream>>>(xx, wr16, rbuf);

  wkv_kernel<<<(BSZ * CDIM) / 256, 256, 0, stream>>>(kbuf, vbuf, rbuf, td, tf,
                                                     ybuf);

  gemm_bf16_kernel<<<gemmBlocks, 256, 0, stream>>>(ybuf, wo16, out);
}
